// UpConvBlock_47519518163472
// MI455X (gfx1250) — compile-verified, matrix operand order checked by reading, not run on hardware
//
#include <hip/hip_runtime.h>
#include <cstddef>

typedef _Float16 h16;
typedef __attribute__((ext_vector_type(8)))  _Float16 v8h;
typedef __attribute__((ext_vector_type(16))) _Float16 v16h;
typedef __attribute__((ext_vector_type(8)))  float v8f;

#define NB   4     // batch
#define CIN  256
#define CO   128
#define HIN  64
#define WIN  64
#define HO   128
#define WO   128
#define XTP  66    // padded xt height/width (halo 1)
#define YTP  130   // padded yt height/width (halo 1)

static __device__ __forceinline__ v16h cat16(v8h lo, v8h hi) {
  return __builtin_shufflevector(lo, hi, 0,1,2,3,4,5,6,7,8,9,10,11,12,13,14,15);
}
// base must already include the per-lane kh2*8 term; off is a compile-time constant
static __device__ __forceinline__ v16h ldfrag(const h16* base, int off) {
  v8h lo = *(const v8h*)(base + off);
  v8h hi = *(const v8h*)(base + off + 16);
  return cat16(lo, hi);
}

// ---------- pack kernels ----------

// x (B,CIN,HIN,WIN) f32 -> xt (B,XTP,XTP,CIN) f16, interior at [1..64]
__global__ void pack_x_kernel(const float* __restrict__ x, h16* __restrict__ xt) {
  int idx = blockIdx.x * blockDim.x + threadIdx.x;          // over NB*CIN*HIN*WIN
  int w  = idx & 63;
  int hh = (idx >> 6) & 63;
  int ci = (idx >> 12) & 255;
  int b  = idx >> 20;
  xt[((size_t)((b * XTP + hh + 1) * XTP + w + 1)) * CIN + ci] = (h16)x[idx];
}

// w1 (CIN,CO,4,4) f32 -> w1h [kh][kw][CO][CIN] f16
__global__ void pack_w1_kernel(const float* __restrict__ w1, h16* __restrict__ w1h) {
  int idx = blockIdx.x * blockDim.x + threadIdx.x;          // over CIN*CO*16
  int kw = idx & 3;
  int kh = (idx >> 2) & 3;
  int co = (idx >> 4) & 127;
  int ci = idx >> 11;
  w1h[((size_t)((kh * 4 + kw) * CO + co)) * CIN + ci] = (h16)w1[idx];
}

// w2 (CO,CO,3,3) f32 -> w2h [tap][O][C] f16 with wk[o,c,i,j] = w2[c,o,2-i,2-j], tap=i*3+j
__global__ void pack_w2_kernel(const float* __restrict__ w2, h16* __restrict__ w2h) {
  int idx = blockIdx.x * blockDim.x + threadIdx.x;          // over CO*CO*9
  int jb = idx % 3;
  int ia = (idx / 3) % 3;
  int o  = (idx / 9) % CO;
  int c  = idx / (9 * CO);
  int tap = (2 - ia) * 3 + (2 - jb);
  w2h[((size_t)(tap * CO + o)) * CO + c] = (h16)w2[idx];
}

// ---------- per-pixel PAC Gaussian kernel: k_all[b][tap][h][w] ----------
__global__ void kguide_kernel(const float* __restrict__ g, float* __restrict__ k_all) {
  int t = blockIdx.x * blockDim.x + threadIdx.x;            // over NB*HO*WO
  int w  = t & 127;
  int hh = (t >> 7) & 127;
  int b  = t >> 14;

  int   off[9];
  float msk[9];
#pragma unroll
  for (int i = 0; i < 3; ++i) {
#pragma unroll
    for (int j = 0; j < 3; ++j) {
      int hn = hh + i - 1, wn = w + j - 1;
      bool ok = (hn >= 0) && (hn < HO) && (wn >= 0) && (wn < WO);
      off[i * 3 + j] = ok ? (hn * WO + wn) : 0;
      msk[i * 3 + j] = ok ? 1.0f : 0.0f;
    }
  }
  float s[9];
#pragma unroll
  for (int q = 0; q < 9; ++q) s[q] = 0.0f;
  for (int d = 0; d < CO; ++d) {
    const float* gp = g + ((size_t)(b * CO + d)) * HO * WO;
    float gc = gp[hh * WO + w];
#pragma unroll
    for (int q = 0; q < 9; ++q) {
      float df = msk[q] * gp[off[q]] - gc;
      s[q] += df * df;
    }
  }
#pragma unroll
  for (int tp = 0; tp < 9; ++tp)
    k_all[((size_t)((b * 9 + tp) * HO + hh)) * WO + w] = expf(-0.5f * s[tp]);
}

// ---------- stage 1: conv-transpose 4x4 s2 p1, 4-parity implicit GEMM ----------
// 32(M=cout) x 32(N=pixels) macro-tile per wave: 2 A-frags x 2 B-frags -> 4 acc,
// 1 fragment load per WMMA. One base pointer per operand; all tap/K offsets are
// compile-time constants folded into the load IOFFSET (dh = po-th, dw = qo-tw,
// tap index = (kh0*4+kw0) + 8*th + 2*tw).
__global__ void __launch_bounds__(128)
stage1_wmma_kernel(const h16* __restrict__ xt, const h16* __restrict__ w1h,
                   const float* __restrict__ b1, h16* __restrict__ yt) {
  int wave = (blockIdx.x << 2) | (threadIdx.x >> 5);
  int lane = threadIdx.x & 31;
  int n    = lane & 15;        // pixel (N) index within frag; also A-fragment row M
  int kh2  = lane >> 4;        // K-half select per ISA 16-bit A/B layout

  int t   = wave;
  int cT  = t & 1;  t >>= 1;   // 2 pixel 32-chunks
  int pr  = t & 63; t >>= 6;
  int coT = t & 3;  t >>= 2;   // 4 cout 32-chunks
  int par = t & 3;  t >>= 2;
  int b   = t;
  int po = par >> 1, qo = par & 1;
  int co0 = coT * 32;
  int c0p = cT * 32;
  int kh0 = (po + 1) & 1, kw0 = (qo + 1) & 1;

  // anchored base pointers (include per-lane kh2*8 K-offset)
  const h16* bbase0 = xt + ((size_t)((b * XTP + pr + po + 1) * XTP + c0p + n + qo + 1)) * CIN + kh2 * 8;
  const h16* bbase1 = bbase0 + 16 * CIN;
  const h16* abase0 = w1h + ((size_t)((kh0 * 4 + kw0) * CO + co0 + n)) * CIN + kh2 * 8;
  const h16* abase1 = abase0 + 16 * CIN;

  v8f acc[2][2];               // [mi][ni]
#pragma unroll
  for (int mi = 0; mi < 2; ++mi)
#pragma unroll
    for (int ni = 0; ni < 2; ++ni) acc[mi][ni] = (v8f){};

#pragma unroll
  for (int th = 0; th < 2; ++th) {
#pragma unroll
    for (int tw = 0; tw < 2; ++tw) {
      const int aoffT = (8 * th + 2 * tw) * CO * CIN;   // constant
      const int boffT = (-th * XTP - tw) * CIN;         // constant
#pragma unroll
      for (int k0 = 0; k0 < CIN; k0 += 32) {
        v16h A0 = ldfrag(abase0, aoffT + k0);
        v16h A1 = ldfrag(abase1, aoffT + k0);
        v16h B0 = ldfrag(bbase0, boffT + k0);
        v16h B1 = ldfrag(bbase1, boffT + k0);
        acc[0][0] = __builtin_amdgcn_wmma_f32_16x16x32_f16(false, A0, false, B0, (short)0, acc[0][0], false, false);
        acc[0][1] = __builtin_amdgcn_wmma_f32_16x16x32_f16(false, A0, false, B1, (short)0, acc[0][1], false, false);
        acc[1][0] = __builtin_amdgcn_wmma_f32_16x16x32_f16(false, A1, false, B0, (short)0, acc[1][0], false, false);
        acc[1][1] = __builtin_amdgcn_wmma_f32_16x16x32_f16(false, A1, false, B1, (short)0, acc[1][1], false, false);
      }
    }
  }

  int oh = 2 * pr + po;
#pragma unroll
  for (int mi = 0; mi < 2; ++mi) {
#pragma unroll
    for (int v = 0; v < 8; ++v) {            // C layout: M = v + 8*kh2, N = n
      int co = co0 + mi * 16 + v + 8 * kh2;
      float bb = b1[co];
#pragma unroll
      for (int ni = 0; ni < 2; ++ni) {
        int ow = 2 * (c0p + ni * 16 + n) + qo;
        yt[((size_t)((b * YTP + oh + 1) * YTP + ow + 1)) * CO + co] = (h16)(acc[mi][ni][v] + bb);
      }
    }
  }
}

// ---------- stage 2: PAC 3x3 as 9-tap GEMM, per-pixel scale applied post-GEMM ----------
// 32(M=out-ch) x 16(N=pixels) macro-tile: each B-fragment (y) feeds 2 WMMAs.
// One base pointer per operand; all tap/K offsets compile-time constants.
// 2x9 accumulators; k_tap folded in the fp32 epilogue. Halo-padded yt: OOB taps exact 0.
__global__ void __launch_bounds__(128)
stage2_wmma_kernel(const h16* __restrict__ yt, const h16* __restrict__ w2h,
                   const float* __restrict__ b2, const float* __restrict__ k_all,
                   float* __restrict__ out) {
  int wave = (blockIdx.x << 2) | (threadIdx.x >> 5);
  int lane = threadIdx.x & 31;
  int n    = lane & 15;
  int kh2  = lane >> 4;

  int t  = wave;
  int wT = t & 7;   t >>= 3;
  int hh = t & 127; t >>= 7;
  int oT = t & 3;   t >>= 2;   // 4 out-channel 32-chunks
  int b  = t;
  int o0 = oT * 32, w0 = wT * 16;
  int w = w0 + n;

  // anchored base pointers (padded coords absorb the -1 halo; include kh2*8)
  const h16* bbase  = yt + ((size_t)((b * YTP + hh) * YTP + w)) * CO + kh2 * 8;
  const h16* abase0 = w2h + ((size_t)(o0 + n)) * CO + kh2 * 8;
  const h16* abase1 = abase0 + 16 * CO;

  v8f acc[2][9];
#pragma unroll
  for (int mi = 0; mi < 2; ++mi)
#pragma unroll
    for (int q = 0; q < 9; ++q) acc[mi][q] = (v8f){};

#pragma unroll
  for (int i = 0; i < 3; ++i) {
#pragma unroll
    for (int j = 0; j < 3; ++j) {
      const int tap  = i * 3 + j;
      const int boffT = (i * YTP + j) * CO;             // constant
      const int aoffT = tap * CO * CO;                  // constant
#pragma unroll
      for (int k0 = 0; k0 < CO; k0 += 32) {
        v16h B  = ldfrag(bbase,  boffT + k0);
        v16h A0 = ldfrag(abase0, aoffT + k0);
        v16h A1 = ldfrag(abase1, aoffT + k0);
        acc[0][tap] = __builtin_amdgcn_wmma_f32_16x16x32_f16(false, A0, false, B, (short)0, acc[0][tap], false, false);
        acc[1][tap] = __builtin_amdgcn_wmma_f32_16x16x32_f16(false, A1, false, B, (short)0, acc[1][tap], false, false);
      }
    }
  }

  float kv[9];
#pragma unroll
  for (int q = 0; q < 9; ++q)
    kv[q] = k_all[((size_t)((b * 9 + q) * HO + hh)) * WO + w];

#pragma unroll
  for (int mi = 0; mi < 2; ++mi) {
#pragma unroll
    for (int v = 0; v < 8; ++v) {
      int o = o0 + mi * 16 + v + 8 * kh2;
      float r = b2[o];
#pragma unroll
      for (int q = 0; q < 9; ++q) r += kv[q] * acc[mi][q][v];
      out[((size_t)((b * CO + o) * HO + hh)) * WO + w] = r;
    }
  }
}

// ---------- launch ----------
extern "C" void kernel_launch(void* const* d_in, const int* in_sizes, int n_in,
                              void* d_out, int out_size, void* d_ws, size_t ws_size,
                              hipStream_t stream) {
  (void)in_sizes; (void)n_in; (void)out_size; (void)ws_size;
  const float* x     = (const float*)d_in[0];
  const float* guide = (const float*)d_in[1];
  const float* w1    = (const float*)d_in[2];
  const float* b1    = (const float*)d_in[3];
  const float* w2    = (const float*)d_in[4];
  const float* b2    = (const float*)d_in[5];
  float* out = (float*)d_out;

  char* ws = (char*)d_ws;
  // workspace layout (all 256B aligned)
  const size_t XT_BYTES = (size_t)NB * XTP * XTP * CIN * 2;   //  8,921,088
  const size_t YT_BYTES = (size_t)NB * YTP * YTP * CO * 2;    // 17,305,600
  h16*   xt    = (h16*)(ws + 0);
  h16*   w1h   = (h16*)(ws + 8921088);           //  1,048,576 B
  h16*   w2h   = (h16*)(ws + 9969664);           //    294,912 B
  h16*   yt    = (h16*)(ws + 10264576);
  float* k_all = (float*)(ws + 27570176);        //  2,359,296 B

  // zero halos (graph-capture-safe async memsets; ~26MB, trivial vs 23.3 TB/s)
  hipMemsetAsync(xt, 0, XT_BYTES, stream);
  hipMemsetAsync(yt, 0, YT_BYTES, stream);

  pack_x_kernel <<<(NB * CIN * HIN * WIN) / 256, 256, 0, stream>>>(x, xt);
  pack_w1_kernel<<<(CIN * CO * 16) / 256,        256, 0, stream>>>(w1, w1h);
  pack_w2_kernel<<<(CO * CO * 9) / 256,          256, 0, stream>>>(w2, w2h);
  kguide_kernel <<<(NB * HO * WO) / 256,         256, 0, stream>>>(guide, k_all);

  // stage 1: 4(b) * 4(parity) * 4(coT32) * 64(row) * 2(cT32) = 8192 wave-tiles, 4 waves/block
  stage1_wmma_kernel<<<2048, 128, 0, stream>>>(xt, w1h, b1, yt);

  // stage 2: 4(b) * 4(oT32) * 128(h) * 8(wT) = 16384 wave-tiles, 4 waves/block
  stage2_wmma_kernel<<<4096, 128, 0, stream>>>(yt, w2h, b2, k_all, out);
}
